// MultiHeadSelfAttention_69002944577554
// MI455X (gfx1250) — compile-verified, matrix operand order checked by reading, not run on hardware
//
#include <hip/hip_runtime.h>
#include <hip/hip_bf16.h>

// ---------------------------------------------------------------------------
// Multi-head self-attention, B=2 T=2048 D=1024 H=16 HD=64, bf16 WMMA pipeline
// Round 3: async-to-LDS double-buffered block-tiled GEMMs (128x64 tiles,
// swizzled LDS), LDS-tiled weight transposes, async-staged flash attention.
// ---------------------------------------------------------------------------

typedef __attribute__((ext_vector_type(16))) __bf16 v16bf;
typedef __attribute__((ext_vector_type(8)))  __bf16 bf16x8;
typedef __attribute__((ext_vector_type(8)))  float  v8f;

#define Bc  2
#define Tc  2048
#define Dc  1024
#define Hc  16
#define HDc 64

__device__ __forceinline__ v8f wmma_bf16(v16bf a, v16bf b, v8f c) {
    return __builtin_amdgcn_wmma_f32_16x16x32_bf16(
        false, a, false, b, (short)0, c, false, false);
}

// Load a 16x32 (MxK) bf16 fragment from a row-major global matrix.
__device__ __forceinline__ v16bf load_frag(const __bf16* __restrict__ base,
                                           int row0, int ld, int k0) {
    const int l  = threadIdx.x & 31;
    const int lm = l & 15;
    const int lh = l >> 4;
    const __bf16* p = base + (size_t)(row0 + lm) * (size_t)ld + (size_t)(k0 + 8 * lh);
    bf16x8 lo = *(const bf16x8*)p;
    bf16x8 hi = *(const bf16x8*)(p + 16);
    v16bf f;
#pragma unroll
    for (int i = 0; i < 8; ++i) { f[i] = lo[i]; f[8 + i] = hi[i]; }
    return f;
}

__device__ __forceinline__ v16bf combine(bf16x8 lo, bf16x8 hi) {
    v16bf f;
#pragma unroll
    for (int i = 0; i < 8; ++i) { f[i] = lo[i]; f[8 + i] = hi[i]; }
    return f;
}

// Wave-relative LDS byte offset of a __shared__ pointer.
__device__ __forceinline__ unsigned lds_off(const void* p) {
    return (unsigned)(unsigned long long)p;
}

// CDNA5 async copy: 16B per lane, global -> LDS, tracked by ASYNCcnt.
__device__ __forceinline__ void async_copy16(unsigned lds_addr, const void* gaddr) {
    asm volatile("global_load_async_to_lds_b128 %0, %1, off"
                 :: "v"(lds_addr), "v"((unsigned long long)gaddr)
                 : "memory");
}

// ---------------------------------------------------------------------------
// K0a: x -> bf16 (already coalesced)
// ---------------------------------------------------------------------------
__global__ __launch_bounds__(256) void cast_kernel(
    const float* __restrict__ src, __bf16* __restrict__ dst, int n) {
    const int i = blockIdx.x * 256 + threadIdx.x;
    if (i < n) dst[i] = (__bf16)src[i];
}

// ---------------------------------------------------------------------------
// K0b: LDS-tiled transpose + cast: dst[n][k] = bf16(src[k][n]); src is RxC.
// One 32x32 tile per block, coalesced on both sides.
// ---------------------------------------------------------------------------
__global__ __launch_bounds__(256) void transpose_cast_kernel(
    const float* __restrict__ src, __bf16* __restrict__ dst, int R, int C) {
    __shared__ float tile[32][33];
    const int tc = threadIdx.x & 31;
    const int tr = threadIdx.x >> 5;
    const int nb = blockIdx.x % (C >> 5);
    const int kb = blockIdx.x / (C >> 5);
#pragma unroll
    for (int rr = 0; rr < 4; ++rr) {
        const int k = kb * 32 + tr + 8 * rr;
        const int n = nb * 32 + tc;
        tile[tc][tr + 8 * rr] = src[(size_t)k * C + n];   // tile[n_loc][k_loc]
    }
    __syncthreads();
#pragma unroll
    for (int rr = 0; rr < 4; ++rr) {
        const int n = nb * 32 + tr + 8 * rr;
        const int k = kb * 32 + tc;
        dst[(size_t)n * R + k] = (__bf16)tile[tr + 8 * rr][tc];
    }
}

// ---------------------------------------------------------------------------
// Block-tiled GEMM core: 256 threads / 8 waves, 128(M) x 64(N) tile, K-step 32.
// A/B K-slices staged in LDS via double-buffered async copies; 16B chunk index
// XOR-swizzled by (row&3) to avoid bank conflicts (same swizzle on write+read).
// Each wave owns a 16-row slice and all four 16-col groups: 4 WMMAs / K-step.
// Result: c[4] accumulators (v8f each).
// ---------------------------------------------------------------------------
#define GEMM_TILE_BODY(Aptr, Bptr, m0, n0, KDIM)                                   \
    __shared__ __bf16 Al[2][128 * 32];  /* 8KB per buffer */                       \
    __shared__ __bf16 Bl[2][64 * 32];   /* 4KB per buffer */                       \
    const int tid  = threadIdx.x;                                                  \
    const int wave = tid >> 5;                                                     \
    const int l    = tid & 31;                                                     \
    const int lm   = l & 15;                                                       \
    const int lh   = l >> 4;                                                       \
    const int arow0 = tid >> 2, acc0 = tid & 3;            /* A chunk 0 */         \
    const int arow1 = (tid + 256) >> 2, acc1 = tid & 3;    /* A chunk 1 */         \
    const int brow  = tid >> 2, bcc = tid & 3;             /* B chunk   */         \
    const unsigned albase[2] = {lds_off(&Al[0][0]), lds_off(&Al[1][0])};           \
    const unsigned blbase[2] = {lds_off(&Bl[0][0]), lds_off(&Bl[1][0])};           \
    v8f c[4] = {v8f{}, v8f{}, v8f{}, v8f{}};                                       \
    /* prime buffer 0 with K-step 0 */                                             \
    async_copy16(albase[0] + arow0 * 64 + ((acc0 ^ (arow0 & 3)) << 4),             \
                 (const char*)(Aptr + (size_t)(m0 + arow0) * KDIM) + acc0 * 16);   \
    async_copy16(albase[0] + arow1 * 64 + ((acc1 ^ (arow1 & 3)) << 4),             \
                 (const char*)(Aptr + (size_t)(m0 + arow1) * KDIM) + acc1 * 16);   \
    async_copy16(blbase[0] + brow * 64 + ((bcc ^ (brow & 3)) << 4),                \
                 (const char*)(Bptr + (size_t)(n0 + brow) * KDIM) + bcc * 16);     \
    const int nk = KDIM / 32;                                                      \
    for (int ks = 0; ks < nk; ++ks) {                                              \
        const int cur = ks & 1;                                                    \
        if (ks + 1 < nk) {                                                         \
            const int kb = (ks + 1) * 32;                                          \
            const int nxt = cur ^ 1;                                               \
            async_copy16(albase[nxt] + arow0 * 64 + ((acc0 ^ (arow0 & 3)) << 4),   \
                         (const char*)(Aptr + (size_t)(m0 + arow0) * KDIM + kb) +  \
                             acc0 * 16);                                           \
            async_copy16(albase[nxt] + arow1 * 64 + ((acc1 ^ (arow1 & 3)) << 4),   \
                         (const char*)(Aptr + (size_t)(m0 + arow1) * KDIM + kb) +  \
                             acc1 * 16);                                           \
            async_copy16(blbase[nxt] + brow * 64 + ((bcc ^ (brow & 3)) << 4),      \
                         (const char*)(Bptr + (size_t)(n0 + brow) * KDIM + kb) +   \
                             bcc * 16);                                            \
            asm volatile("s_wait_asynccnt 3" ::: "memory");                        \
        } else {                                                                   \
            asm volatile("s_wait_asynccnt 0" ::: "memory");                        \
        }                                                                          \
        __syncthreads();                                                           \
        const int ar = wave * 16 + lm;                                             \
        bf16x8 alo = *(const bf16x8*)&Al[cur][ar * 32 + ((lh ^ (ar & 3)) << 3)];   \
        bf16x8 ahi =                                                               \
            *(const bf16x8*)&Al[cur][ar * 32 + (((2 + lh) ^ (ar & 3)) << 3)];      \
        const v16bf af = combine(alo, ahi);                                        \
        _Pragma("unroll") for (int g = 0; g < 4; ++g) {                            \
            const int br = g * 16 + lm;                                            \
            bf16x8 blo =                                                           \
                *(const bf16x8*)&Bl[cur][br * 32 + ((lh ^ (br & 3)) << 3)];        \
            bf16x8 bhi =                                                           \
                *(const bf16x8*)&Bl[cur][br * 32 + (((2 + lh) ^ (br & 3)) << 3)];  \
            c[g] = wmma_bf16(af, combine(blo, bhi), c[g]);                         \
        }                                                                          \
        __syncthreads();                                                           \
    }

// ---------------------------------------------------------------------------
// K1: QKV projection (4096x3072x1024): 32 x 48 blocks of 128x64 tiles.
// ---------------------------------------------------------------------------
__global__ __launch_bounds__(256) void gemm_qkv_kernel(
    const __bf16* __restrict__ xb, const __bf16* __restrict__ wqkvT,
    __bf16* __restrict__ Qh, __bf16* __restrict__ Kh, __bf16* __restrict__ Vt) {
    const int m0 = (blockIdx.x / 48) * 128;
    const int n0 = (blockIdx.x % 48) * 64;
    GEMM_TILE_BODY(xb, wqkvT, m0, n0, Dc)

#pragma unroll
    for (int g = 0; g < 4; ++g) {
        const int n = n0 + 16 * g + lm;
#pragma unroll
        for (int r = 0; r < 8; ++r) {
            const int m = m0 + wave * 16 + r + 8 * lh;
            const int b = m >> 11;
            const int t = m & (Tc - 1);
            const float v = c[g][r];
            if (n < Dc) {                       // Q (pre-scaled by 1/sqrt(HD))
                const int h = n >> 6, hd = n & 63;
                Qh[((size_t)(b * Hc + h) * Tc + t) * HDc + hd] = (__bf16)(v * 0.125f);
            } else if (n < 2 * Dc) {
                const int n2 = n - Dc, h = n2 >> 6, hd = n2 & 63;
                Kh[((size_t)(b * Hc + h) * Tc + t) * HDc + hd] = (__bf16)v;
            } else {
                const int n2 = n - 2 * Dc, h = n2 >> 6, hd = n2 & 63;
                Vt[((size_t)(b * Hc + h) * HDc + hd) * Tc + t] = (__bf16)v;
            }
        }
    }
}

// ---------------------------------------------------------------------------
// K3: output projection (4096x1024x1024): 32 x 16 blocks of 128x64 tiles.
// ---------------------------------------------------------------------------
__global__ __launch_bounds__(256) void gemm_out_kernel(
    const __bf16* __restrict__ Ao, const __bf16* __restrict__ woT,
    float* __restrict__ out) {
    const int m0 = (blockIdx.x >> 4) * 128;
    const int n0 = (blockIdx.x & 15) * 64;
    GEMM_TILE_BODY(Ao, woT, m0, n0, Dc)

#pragma unroll
    for (int g = 0; g < 4; ++g) {
        const int n = n0 + 16 * g + lm;
#pragma unroll
        for (int r = 0; r < 8; ++r) {
            const int m = m0 + wave * 16 + r + 8 * lh;
            out[(size_t)m * Dc + n] = c[g][r];
        }
    }
}

// ---------------------------------------------------------------------------
// K2: flash attention. 8 waves/block share one (b,h); K/V chunks staged in
// LDS via async-to-LDS copies, double-buffered. One wave per 16-query tile.
// ---------------------------------------------------------------------------
__global__ __launch_bounds__(256) void attn_kernel(
    const __bf16* __restrict__ Qh, const __bf16* __restrict__ Kh,
    const __bf16* __restrict__ Vt, const int* __restrict__ mask,
    __bf16* __restrict__ Ao) {
    __shared__ __bf16 Kl[2][32 * HDc];   // [key][d]   4KB per buffer
    __shared__ __bf16 Vl[2][HDc * 32];   // [d][key]   4KB per buffer
    __shared__ __bf16 Pl[8][16 * 32];    // P staging, 1KB per wave

    const int tid  = threadIdx.x;
    const int wave = tid >> 5;
    const int l    = tid & 31;
    const int lm   = l & 15;
    const int lh   = l >> 4;

    const int bh = (blockIdx.x * 8) >> 7;          // uniform across the block
    const int qt = (blockIdx.x & 15) * 8 + wave;
    const int b  = bh >> 4;
    const int h  = bh & 15;
    const int q0 = qt * 16;

    const __bf16* Qp = Qh + (size_t)bh * Tc * HDc;
    const __bf16* Kp = Kh + (size_t)bh * Tc * HDc;
    const __bf16* Vp = Vt + (size_t)bh * HDc * Tc;
    const int*    mp = mask + b * Tc;

    const v16bf qa0 = load_frag(Qp, q0, HDc, 0);
    const v16bf qa1 = load_frag(Qp, q0, HDc, 32);

    const unsigned kl0 = lds_off(&Kl[0][0]) + (unsigned)tid * 16;
    const unsigned kl1 = lds_off(&Kl[1][0]) + (unsigned)tid * 16;
    const int vrow = tid >> 2;            // dim 0..63
    const int vc   = tid & 3;             // 16B chunk within a 32-key row
    const unsigned vl0 = lds_off(&Vl[0][0]) + (unsigned)(vrow * 64 + vc * 16);
    const unsigned vl1 = lds_off(&Vl[1][0]) + (unsigned)(vrow * 64 + vc * 16);

    async_copy16(kl0, (const char*)(Kp) + (size_t)tid * 16);
    async_copy16(vl0, (const char*)(Vp + (size_t)vrow * Tc) + vc * 16);

    v8f o[4] = {v8f{}, v8f{}, v8f{}, v8f{}};
    float rm[8], rs[8];
#pragma unroll
    for (int r = 0; r < 8; ++r) { rm[r] = -__builtin_inff(); rs[r] = 0.f; }

    const int niter = Tc / 32;
    for (int it = 0; it < niter; ++it) {
        const int kv  = it * 32;
        const int cur = it & 1;

        if (it + 1 < niter) {
            const int kn = kv + 32;
            async_copy16(cur ? kl0 : kl1,
                         (const char*)(Kp + (size_t)kn * HDc) + (size_t)tid * 16);
            async_copy16(cur ? vl0 : vl1,
                         (const char*)(Vp + (size_t)vrow * Tc + kn) + vc * 16);
            asm volatile("s_wait_asynccnt 2" ::: "memory");
        } else {
            asm volatile("s_wait_asynccnt 0" ::: "memory");
        }
        __syncthreads();

        // ---- S = Q * K^T for 32 keys from LDS (B-frag: column = key)
        v8f s0 = {}, s1 = {};
#pragma unroll
        for (int kt = 0; kt < 2; ++kt) {
            v8f* s = kt ? &s1 : &s0;
#pragma unroll
            for (int d0 = 0; d0 < 64; d0 += 32) {
                bf16x8 blo = *(const bf16x8*)&Kl[cur][(kt * 16 + lm) * HDc + d0 + 8 * lh];
                bf16x8 bhi = *(const bf16x8*)&Kl[cur][(kt * 16 + lm) * HDc + d0 + 16 + 8 * lh];
                *s = wmma_bf16(d0 ? qa1 : qa0, combine(blo, bhi), *s);
            }
        }

        const float mk0 = (mp[kv + lm]      != 0) ? 0.f : -__builtin_inff();
        const float mk1 = (mp[kv + 16 + lm] != 0) ? 0.f : -__builtin_inff();

        // ---- online softmax over this 32-key chunk
        float p0[8], p1[8];
#pragma unroll
        for (int r = 0; r < 8; ++r) {
            const float a0 = s0[r] + mk0;
            const float a1 = s1[r] + mk1;
            float cm = fmaxf(a0, a1);
            cm = fmaxf(cm, __shfl_xor(cm, 1, 32));
            cm = fmaxf(cm, __shfl_xor(cm, 2, 32));
            cm = fmaxf(cm, __shfl_xor(cm, 4, 32));
            cm = fmaxf(cm, __shfl_xor(cm, 8, 32));
            const float nm    = fmaxf(rm[r], cm);
            const float alpha = __expf(rm[r] - nm);
            p0[r] = __expf(a0 - nm);
            p1[r] = __expf(a1 - nm);
            float ps = p0[r] + p1[r];
            ps += __shfl_xor(ps, 1, 32);
            ps += __shfl_xor(ps, 2, 32);
            ps += __shfl_xor(ps, 4, 32);
            ps += __shfl_xor(ps, 8, 32);
            rs[r] = rs[r] * alpha + ps;
            rm[r] = nm;
            o[0][r] *= alpha; o[1][r] *= alpha; o[2][r] *= alpha; o[3][r] *= alpha;
        }

        // ---- C-layout P -> LDS -> A-layout fragment
#pragma unroll
        for (int r = 0; r < 8; ++r) {
            Pl[wave][(r + 8 * lh) * 32 + lm]      = (__bf16)p0[r];
            Pl[wave][(r + 8 * lh) * 32 + 16 + lm] = (__bf16)p1[r];
        }
        asm volatile("s_wait_dscnt 0" ::: "memory");
        __builtin_amdgcn_wave_barrier();
        bf16x8 plo = *(const bf16x8*)&Pl[wave][lm * 32 + 8 * lh];
        bf16x8 phi = *(const bf16x8*)&Pl[wave][lm * 32 + 16 + 8 * lh];
        const v16bf pa = combine(plo, phi);
        __builtin_amdgcn_wave_barrier();

        // ---- O += P(16x32) * V(32x64) from LDS (B-frag: column = dim)
#pragma unroll
        for (int g = 0; g < 4; ++g) {
            bf16x8 vlo = *(const bf16x8*)&Vl[cur][(16 * g + lm) * 32 + 8 * lh];
            bf16x8 vhi = *(const bf16x8*)&Vl[cur][(16 * g + lm) * 32 + 16 + 8 * lh];
            o[g] = wmma_bf16(pa, combine(vlo, vhi), o[g]);
        }

        __syncthreads();   // all reads of buf[cur] done before it is overwritten
    }

    // ---- normalize, scatter to [b, t, h*64+d]
#pragma unroll
    for (int g = 0; g < 4; ++g) {
        const int d = 16 * g + lm;
#pragma unroll
        for (int r = 0; r < 8; ++r) {
            const int t = q0 + r + 8 * lh;
            Ao[((size_t)(b * Tc + t)) * Dc + h * HDc + d] = (__bf16)(o[g][r] / rs[r]);
        }
    }
}

// ---------------------------------------------------------------------------
// Launcher
// ---------------------------------------------------------------------------
extern "C" void kernel_launch(void* const* d_in, const int* in_sizes, int n_in,
                              void* d_out, int out_size, void* d_ws, size_t ws_size,
                              hipStream_t stream) {
    const float* x     = (const float*)d_in[0];
    const int*   amask = (const int*)d_in[1];
    const float* wqkv  = (const float*)d_in[2];
    const float* wout  = (const float*)d_in[3];
    float*       out   = (float*)d_out;

    char* ws = (char*)d_ws;
    size_t off = 0;
    __bf16* xb    = (__bf16*)(ws + off); off += (size_t)Bc * Tc * Dc * 2;
    __bf16* wqkvT = (__bf16*)(ws + off); off += (size_t)3 * Dc * Dc * 2;
    __bf16* woT   = (__bf16*)(ws + off); off += (size_t)Dc * Dc * 2;
    __bf16* Qh    = (__bf16*)(ws + off); off += (size_t)Bc * Hc * Tc * HDc * 2;
    __bf16* Kh    = (__bf16*)(ws + off); off += (size_t)Bc * Hc * Tc * HDc * 2;
    __bf16* Vt    = (__bf16*)(ws + off); off += (size_t)Bc * Hc * HDc * Tc * 2;
    __bf16* Ao    = (__bf16*)(ws + off); off += (size_t)Bc * Tc * Dc * 2;

    cast_kernel<<<(Bc * Tc * Dc + 255) / 256, 256, 0, stream>>>(x, xb, Bc * Tc * Dc);
    transpose_cast_kernel<<<(Dc / 32) * (3 * Dc / 32), 256, 0, stream>>>(
        wqkv, wqkvT, Dc, 3 * Dc);
    transpose_cast_kernel<<<(Dc / 32) * (Dc / 32), 256, 0, stream>>>(
        wout, woT, Dc, Dc);

    gemm_qkv_kernel<<<32 * 48, 256, 0, stream>>>(xb, wqkvT, Qh, Kh, Vt);
    attn_kernel<<<(Bc * Hc * (Tc / 16)) / 8, 256, 0, stream>>>(Qh, Kh, Vt, amask, Ao);
    gemm_out_kernel<<<32 * 16, 256, 0, stream>>>(Ao, woT, out);
}
